// PreprocessPositions_45629732552735
// MI455X (gfx1250) — matrix operand order repaired and checked by reading, not compiled, not run on hardware
//
#include <hip/hip_runtime.h>

typedef __attribute__((ext_vector_type(16))) _Float16 v16h;
typedef __attribute__((ext_vector_type(8)))  float    v8f;

#define B_ 8
#define N_ 512
#define K_ 64
#define H_ 32
#define E_ 768

#define INV_SQRT_2PI 0.39894228040143268f
#define NEG_HALF_LOG2E (-0.72134752044448170f)   // -0.5 * log2(e)

__device__ __forceinline__ float gelu_exact(float x) {
    return 0.5f * x * (1.0f + erff(x * 0.70710678118654752f));
}

__device__ __forceinline__ unsigned pack2h(float a, float b) {
    _Float16 ha = (_Float16)a, hb = (_Float16)b;
    unsigned short ua = __builtin_bit_cast(unsigned short, ha);
    unsigned short ub = __builtin_bit_cast(unsigned short, hb);
    return (unsigned)ua | ((unsigned)ub << 16);
}

__device__ __forceinline__ v8f v8f_add(v8f a, v8f b) {
    v8f r;
#pragma unroll
    for (int q = 0; q < 8; ++q) r[q] = a[q] + b[q];   // lowers to v_pk_add_f32 pairs
    return r;
}

// SWAPX16: group-of-32 swizzle, xor_mask=0x10, and_mask=0x1f -> exchange lane L <-> L^16
#define SWZ_SWAPX16 0x401F

struct SharedK1 {
    v16h a1[8 * 32];   // coef-folded W1^T A-fragments, lane-swizzled: [frag][lane] (32B aligned)
    v16h a2[4 * 32];   // W2^T A-fragments, lane-swizzled
    float feat[64];
};

__global__ __launch_bounds__(128, 4)   // cap at 256 VGPRs: no s_set_vgpr_msb, 4 waves/SIMD
void attn_bias_kernel(const float* __restrict__ pos, const unsigned char* __restrict__ mask,
                      const float* __restrict__ means, const float* __restrict__ stds,
                      const float* __restrict__ W1, const float* __restrict__ b1,
                      const float* __restrict__ W2, const float* __restrict__ b2,
                      float* __restrict__ attn, float* __restrict__ featWs)
{
    __shared__ SharedK1 s;
    const int t    = threadIdx.x;
    const int lane = t & 31;
    const int wave = t >> 5;
    const int lo   = lane & 15;
    const bool beta = (lane >= 16);
    const int hiA  = beta ? 8 : 0;    // D row offset for upper half
    const int hiB  = beta ? 16 : 0;   // B-fragment K offset for upper half
    const int b    = blockIdx.x >> 9;
    const int i    = blockIdx.x & (N_ - 1);

    // ---- one-time: build lane-swizzled A-fragments. W1^T rows are pre-scaled by the
    //      gaussian coefficient 1/(sqrt(2pi)*std_k), so the inner loop feeds raw exp() values.
    if (t < 64) s.feat[t] = 0.0f;
    if (wave == 0) {
#pragma unroll
        for (int f = 0; f < 8; ++f) {          // f = rb*2 + kb ; rows k2 = rb*16+lo, K = RBF k
            int rb = f >> 1, kb = f & 1;
            v16h fr;
#pragma unroll
            for (int e = 0; e < 16; ++e) {
                int k = hiB + e + kb * 32;
                float coef = INV_SQRT_2PI / (fabsf(stds[k]) + 0.01f);
                fr[e] = (_Float16)(W1[(size_t)k * 64 + rb * 16 + lo] * coef);  // coef*W1^T[k2][k]
            }
            s.a1[f * 32 + lane] = fr;
        }
    } else if (wave == 1) {
#pragma unroll
        for (int f = 0; f < 4; ++f) {          // f = hb*2 + kb2 ; rows h = hb*16+lo, K = k2
            int hb = f >> 1, kb = f & 1;
            v16h fr;
#pragma unroll
            for (int e = 0; e < 16; ++e) {
                int k2 = hiB + e + kb * 32;
                fr[e] = (_Float16)W2[(size_t)k2 * 32 + hb * 16 + lo]; // W2^T[h][k2]
            }
            s.a2[f * 32 + lane] = fr;
        }
    }
    __syncthreads();

    // ---- per-lane loop constants: this lane's RBF kernels are k = kb*32 + hiB + e ----
    float inv_r[32], nmi_r[32];                    // 1/std ; -mean/std  (z = fma(dist,inv,nmi))
#pragma unroll
    for (int kb = 0; kb < 2; ++kb)
#pragma unroll
        for (int e = 0; e < 16; ++e) {
            int k = kb * 32 + hiB + e;
            float inv = 1.0f / (fabsf(stds[k]) + 0.01f);
            inv_r[kb * 16 + e] = inv;
            nmi_r[kb * 16 + e] = -means[k] * inv;
        }
    float b1r[32];                                 // b1 rows this lane's D1 vgprs touch
#pragma unroll
    for (int rb = 0; rb < 4; ++rb)
#pragma unroll
        for (int v = 0; v < 8; ++v) b1r[rb * 8 + v] = b1[rb * 16 + v + hiA];
    float b2r[16];
#pragma unroll
    for (int hb = 0; hb < 2; ++hb)
#pragma unroll
        for (int v = 0; v < 8; ++v) b2r[hb * 8 + v] = b2[hb * 16 + v + hiA];

    const float pix = pos[((size_t)b * N_ + i) * 3 + 0];
    const float piy = pos[((size_t)b * N_ + i) * 3 + 1];
    const float piz = pos[((size_t)b * N_ + i) * 3 + 2];
    const bool nanb = isnan(pos[(size_t)b * N_ * 3]);

    float fa[32];
#pragma unroll
    for (int q = 0; q < 32; ++q) fa[q] = 0.0f;

    for (int jc = 0; jc < 8; ++jc) {
        // keep the A-fragment LDS loads inside the loop (prevents hoist -> scratch spills)
        asm volatile("" ::: "memory");

        const int j = jc * 64 + wave * 16 + lo;    // this lane's key column
        float dx = pix - pos[((size_t)b * N_ + j) * 3 + 0];
        float dy = piy - pos[((size_t)b * N_ + j) * 3 + 1];
        float dz = piz - pos[((size_t)b * N_ + j) * 3 + 2];
        float dist = sqrtf(dx * dx + dy * dy + dz * dz);

        // ---- raw gaussian exponentials in B-fragment (g^T) layout; coef folded into W1 ----
        v16h g0, g1;
#pragma unroll
        for (int e = 0; e < 16; ++e) {
            float z0 = fmaf(dist, inv_r[e], nmi_r[e]);
            float v0 = exp2f(z0 * z0 * NEG_HALF_LOG2E);
            fa[e] += v0;
            g0[e] = (_Float16)v0;
            float z1 = fmaf(dist, inv_r[16 + e], nmi_r[16 + e]);
            float v1 = exp2f(z1 * z1 * NEG_HALF_LOG2E);
            fa[16 + e] += v1;
            g1[e] = (_Float16)v1;
        }

        // ---- GEMM1: H^T = (coef*W1)^T @ e^T ; independent accumulators, no D->C hazard ----
        unsigned pk[4][4];                         // gelu(H^T) as packed f16 pairs per rb tile
#pragma unroll
        for (int rb = 0; rb < 4; ++rb) {
            v16h aA = s.a1[(rb * 2 + 0) * 32 + lane];
            v16h aB = s.a1[(rb * 2 + 1) * 32 + lane];
            v8f c0 = {}, c1 = {};
            c0 = __builtin_amdgcn_wmma_f32_16x16x32_f16(false, aA, false, g0, (short)0, c0, false, false);
            c1 = __builtin_amdgcn_wmma_f32_16x16x32_f16(false, aB, false, g1, (short)0, c1, false, false);
            v8f c = v8f_add(c0, c1);
            float hx[8];
#pragma unroll
            for (int v = 0; v < 8; ++v) hx[v] = gelu_exact(c[v] + b1r[rb * 8 + v]);
#pragma unroll
            for (int p = 0; p < 4; ++p) pk[rb][p] = pack2h(hx[2 * p], hx[2 * p + 1]);
        }

        // ---- relayout H^T tiles into GEMM2 B-fragments via SWAPX16 lane-half exchange ----
        v16h F2[2];
#pragma unroll
        for (int kb = 0; kb < 2; ++kb) {
            union { v16h h; unsigned u[8]; } f2;
#pragma unroll
            for (int p = 0; p < 4; ++p) {
                unsigned x = (unsigned)__builtin_amdgcn_ds_swizzle((int)pk[2 * kb][p], SWZ_SWAPX16);
                unsigned y = (unsigned)__builtin_amdgcn_ds_swizzle((int)pk[2 * kb + 1][p], SWZ_SWAPX16);
                f2.u[p]     = beta ? y : pk[2 * kb][p];      // elements 0..7
                f2.u[4 + p] = beta ? pk[2 * kb + 1][p] : x;  // elements 8..15
            }
            F2[kb] = f2.h;
        }

        // ---- GEMM2: bias^T = W2^T @ H^T ; lane = col j -> fully coalesced direct stores ----
        const bool keep = (mask[b * N_ + j] != 0);
        const bool over = nanb || !keep;
        const float forced = nanb ? 0.0f : -1000.0f;
#pragma unroll
        for (int hb = 0; hb < 2; ++hb) {
            v16h aA = s.a2[(hb * 2 + 0) * 32 + lane];
            v16h aB = s.a2[(hb * 2 + 1) * 32 + lane];
            v8f c0 = {}, c1 = {};
            c0 = __builtin_amdgcn_wmma_f32_16x16x32_f16(false, aA, false, F2[0], (short)0, c0, false, false);
            c1 = __builtin_amdgcn_wmma_f32_16x16x32_f16(false, aB, false, F2[1], (short)0, c1, false, false);
            v8f c = v8f_add(c0, c1);
#pragma unroll
            for (int v = 0; v < 8; ++v) {
                float val = c[v] + b2r[hb * 8 + v];
                val = over ? forced : val;
                int h = hb * 16 + v + hiA;
                attn[((size_t)(b * H_ + h) * N_ + i) * N_ + j] = val;
            }
        }
    }

    // ---- rescale raw exp sums by the gaussian coef, reduce into feat_sum[b,i,:] ----
#pragma unroll
    for (int kb = 0; kb < 2; ++kb)
#pragma unroll
        for (int e = 0; e < 16; ++e)
            atomicAdd(&s.feat[kb * 32 + hiB + e],
                      fa[kb * 16 + e] * (INV_SQRT_2PI * inv_r[kb * 16 + e]));
    __syncthreads();
    if (t < 64) featWs[((size_t)b * N_ + i) * 64 + t] = s.feat[t];
}

__global__ __launch_bounds__(128)
void node_feature_kernel(const float* __restrict__ pos, const float* __restrict__ Wn,
                         const float* __restrict__ bn, const float* __restrict__ featWs,
                         float* __restrict__ nodeOut)
{
    __shared__ _Float16 sF[16 * 72];  // 16 rows x 64 cols, padded to kill bank conflicts
    const int t    = threadIdx.x;
    const int lane = t & 31;
    const int wave = t >> 5;
    const int lo   = lane & 15;
    const int hiA  = (lane >= 16) ? 8 : 0;
    const int hiB  = (lane >= 16) ? 16 : 0;
    const int r0   = blockIdx.x * 16;          // row tile over B*N rows
    const int b    = blockIdx.x >> 5;
    const bool nanb = isnan(pos[(size_t)b * N_ * 3]);

#pragma unroll
    for (int q = 0; q < 8; ++q) {
        int idx = t * 8 + q;                   // 0..1023
        sF[(idx >> 6) * 72 + (idx & 63)] = (_Float16)featWs[(size_t)r0 * 64 + idx];
    }
    __syncthreads();

    v16h a0, a1;
#pragma unroll
    for (int e = 0; e < 16; ++e) {
        int k0 = hiB + e;
        a0[e] = sF[lo * 72 + k0];
        a1[e] = sF[lo * 72 + k0 + 32];
    }

    for (int ct = wave; ct < E_ / 16; ct += 4) {
        v16h bf0, bf1;
#pragma unroll
        for (int e = 0; e < 16; ++e) {
            int k = hiB + e;
            bf0[e] = (_Float16)Wn[(size_t)k * E_ + ct * 16 + lo];
            bf1[e] = (_Float16)Wn[(size_t)(k + 32) * E_ + ct * 16 + lo];
        }
        v8f c0 = {}, c1 = {};
        c0 = __builtin_amdgcn_wmma_f32_16x16x32_f16(false, a0, false, bf0, (short)0, c0, false, false);
        c1 = __builtin_amdgcn_wmma_f32_16x16x32_f16(false, a1, false, bf1, (short)0, c1, false, false);
        float bnv = bn[ct * 16 + lo];
#pragma unroll
        for (int v = 0; v < 8; ++v) {
            int m = v + ((lane >= 16) ? 8 : 0);
            float val = nanb ? 0.0f : (c0[v] + c1[v] + bnv);
            nodeOut[(size_t)(r0 + m) * E_ + ct * 16 + lo] = val;
        }
    }
}

extern "C" void kernel_launch(void* const* d_in, const int* in_sizes, int n_in,
                              void* d_out, int out_size, void* d_ws, size_t ws_size,
                              hipStream_t stream)
{
    (void)in_sizes; (void)n_in; (void)out_size; (void)ws_size;
    const float*         pos   = (const float*)d_in[0];
    const unsigned char* mask  = (const unsigned char*)d_in[1];
    const float*         means = (const float*)d_in[2];
    const float*         stds  = (const float*)d_in[3];
    const float*         W1    = (const float*)d_in[4];
    const float*         b1    = (const float*)d_in[5];
    const float*         W2    = (const float*)d_in[6];
    const float*         b2    = (const float*)d_in[7];
    const float*         Wn    = (const float*)d_in[8];
    const float*         bn    = (const float*)d_in[9];
    float* out    = (float*)d_out;
    float* featWs = (float*)d_ws;   // B*N*64 floats = 1 MB scratch

    attn_bias_kernel<<<B_ * N_, 128, 0, stream>>>(pos, mask, means, stds,
                                                  W1, b1, W2, b2, out, featWs);
    node_feature_kernel<<<(B_ * N_) / 16, 128, 0, stream>>>(
        pos, Wn, bn, featWs, out + (size_t)B_ * H_ * N_ * N_);
}
